// Critic_70282844831799
// MI455X (gfx1250) — matrix operand order repaired and checked by reading, not compiled, not run on hardware
//
#include <hip/hip_runtime.h>
#include <hip/hip_fp16.h>
#include <hip/hip_bf16.h>

#define DEV_INLINE __device__ __forceinline__

typedef _Float16 h16;
typedef __attribute__((ext_vector_type(16))) _Float16 v16h;
typedef __attribute__((ext_vector_type(8)))  _Float16 v8h;
typedef __attribute__((ext_vector_type(8)))  float    v8f;

// ---------------- problem constants ----------------
constexpr int Bc  = 64;          // graphs
constexpr int Lc  = 256;         // nodes per graph
constexpr int Dc  = 256;         // hidden
constexpr int Nn  = Bc * Lc;     // 16384 nodes
constexpr int Ec  = 262144;      // edges
constexpr int INc = 32;          // input dim
constexpr int FFc = 1024;        // FFN dim
constexpr int NHc = 2;           // heads
constexpr int HDc = Dc / NHc;    // 128 head dim
constexpr int ENc = Ec + Nn;     // edges + self loops

// ---------------- WMMA fragment loaders (wave32, 16x16x32 f16) ----------------
// A: 16x32 (MxK). lanes 0-15: M=lane, K 0..15 in v0-7.
//                 lanes 16-31: M=lane-16, K 16..31 in v0-7.
DEV_INLINE v16h load_a_frag(const h16* __restrict__ A, int lda, int row0, int k0) {
  int lane = threadIdx.x & 31;
  int m = lane & 15, hh = lane >> 4;
  const h16* p = A + (size_t)(row0 + m) * lda + k0 + 16 * hh;
  v8h lo = *reinterpret_cast<const v8h*>(p);
  v8h hi = *reinterpret_cast<const v8h*>(p + 8);
  v16h a;
#pragma unroll
  for (int e = 0; e < 8; ++e) { a[e] = lo[e]; a[e + 8] = hi[e]; }
  return a;
}

// B: 32x16 (KxN). lanes 0-15 hold N=lane, K=0..15; lanes 16-31 hold N=lane-16, K=16..31.
// Source is "B transposed" row-major [N x K] (pre-transposed weights, a row-major
// activation used as K^T, or transposed V) -> contiguous 2x b128 per lane.
DEV_INLINE v16h load_b_frag_t(const h16* __restrict__ BT, int ldk, int k0, int n0) {
  int lane = threadIdx.x & 31;
  int n = lane & 15, hh = lane >> 4;
  const h16* p = BT + (size_t)(n0 + n) * ldk + k0 + 16 * hh;
  v8h lo = *reinterpret_cast<const v8h*>(p);
  v8h hi = *reinterpret_cast<const v8h*>(p + 8);
  v16h b;
#pragma unroll
  for (int e = 0; e < 8; ++e) { b[e] = lo[e]; b[e + 8] = hi[e]; }
  return b;
}

DEV_INLINE v8f wmma_f16(v16h a, v16h b, v8f c) {
  // D = A*B + C, f32 accumulate
  return __builtin_amdgcn_wmma_f32_16x16x32_f16(false, a, false, b, (short)0, c,
                                                false, false);
}

DEV_INLINE float wave_sum(float v) {
#pragma unroll
  for (int o = 16; o > 0; o >>= 1) v += __shfl_xor(v, o, 32);
  return v;
}
DEV_INLINE float wave_max(float v) {
#pragma unroll
  for (int o = 16; o > 0; o >>= 1) v = fmaxf(v, __shfl_xor(v, o, 32));
  return v;
}

// order-preserving float<->u32 encoding for atomicMax on floats
DEV_INLINE unsigned fenc(float f) {
  unsigned b = __float_as_uint(f);
  return b ^ (((int)b < 0) ? 0xFFFFFFFFu : 0x80000000u);
}
DEV_INLINE float fdec(unsigned e) {
  unsigned b = (e & 0x80000000u) ? (e ^ 0x80000000u) : ~e;
  return __uint_as_float(b);
}

// ---------------- generic WMMA GEMM ----------------
// C[M,N] = A[M,K](f16) * W[K,N] (W given as WT[N,K] f16).
// Each wave computes a 16x64 strip: one A-fragment feeds 4 WMMAs.
// Plain loop body + unroll-4: the scheduler hoists later iterations' loads
// above earlier WMMAs into distinct registers (no rotation copies), giving
// partial s_wait_loadcnt pipelining.
// mode 0: Cf = acc                          (GAT h)
// mode 1: Ch = f16(acc + bias)              (Q/K)
// mode 2: Ch = f16(relu(acc + bias))        (FFN1)
// mode 3: Cf = acc + bias + R               (pre-layernorm sum)
// mode 4: ChT[col, row] = f16(acc + bias)   (V, stored transposed [N x M],
//                                            per-lane rows contiguous -> b128 store)
__global__ __launch_bounds__(256)
void k_gemm(const h16* __restrict__ A, const h16* __restrict__ WT,
            int M, int N, int K,
            const float* __restrict__ bias, const float* __restrict__ R,
            float* __restrict__ Cf, h16* __restrict__ Ch, int mode)
{
  int wave = threadIdx.x >> 5, lane = threadIdx.x & 31;
  int row0 = blockIdx.x * 128 + wave * 16;
  int n0   = blockIdx.y * 64;
  v8f acc0 = {}, acc1 = {}, acc2 = {}, acc3 = {};
  int ksteps = K >> 5;
#pragma unroll 4
  for (int kt = 0; kt < ksteps; ++kt) {
    int ktn = (kt + 1 < ksteps) ? kt + 1 : kt;   // branch-free prefetch addr
    __builtin_prefetch(A + (size_t)(row0 + (lane & 15)) * K + ktn * 32, 0, 3);
    v16h a  = load_a_frag(A, K, row0, kt * 32);
    v16h b0 = load_b_frag_t(WT, K, kt * 32, n0);
    v16h b1 = load_b_frag_t(WT, K, kt * 32, n0 + 16);
    v16h b2 = load_b_frag_t(WT, K, kt * 32, n0 + 32);
    v16h b3 = load_b_frag_t(WT, K, kt * 32, n0 + 48);
    acc0 = wmma_f16(a, b0, acc0);
    acc1 = wmma_f16(a, b1, acc1);
    acc2 = wmma_f16(a, b2, acc2);
    acc3 = wmma_f16(a, b3, acc3);
  }

  int n = lane & 15, hh = lane >> 4;
  v8f accs[4] = {acc0, acc1, acc2, acc3};
#pragma unroll
  for (int j = 0; j < 4; ++j) {
    int col = n0 + j * 16 + n;
    if (mode == 4) {
      v8h vec;
#pragma unroll
      for (int r = 0; r < 8; ++r) vec[r] = (h16)(accs[j][r] + bias[col]);
      *reinterpret_cast<v8h*>(Ch + (size_t)col * M + row0 + 8 * hh) = vec;
    } else {
#pragma unroll
      for (int r = 0; r < 8; ++r) {
        int row = row0 + r + 8 * hh;
        size_t idx = (size_t)row * N + col;
        float v = accs[j][r];
        if (mode == 0)      { Cf[idx] = v; }
        else if (mode == 1) { v += bias[col]; Ch[idx] = (h16)v; }
        else if (mode == 2) { v += bias[col]; v = v > 0.f ? v : 0.f; Ch[idx] = (h16)v; }
        else                { v += bias[col] + R[idx]; Cf[idx] = v; }
      }
    }
  }
}

// ---------------- small utility kernels ----------------
__global__ void k_cvt(const float* __restrict__ s, h16* __restrict__ d, int n) {
  int i = blockIdx.x * 256 + threadIdx.x;
  if (i < n) d[i] = (h16)s[i];
}
// W[K,N] fp32 -> WT[N,K] f16
__global__ void k_cvt_t(const float* __restrict__ W, h16* __restrict__ WT, int K, int N) {
  int i = blockIdx.x * 256 + threadIdx.x;
  if (i < K * N) { int k = i / N, n = i - k * N; WT[(size_t)n * K + k] = (h16)W[i]; }
}
__global__ void k_fill_u32(unsigned* __restrict__ p, unsigned v, int n) {
  int i = blockIdx.x * 256 + threadIdx.x;
  if (i < n) p[i] = v;
}

// s[i] = H[i,:] . a_src ; t[i] = H[i,:] . a_dst   (one wave per row)
__global__ __launch_bounds__(256)
void k_rowdot(const float* __restrict__ H, const float* __restrict__ a_src,
              const float* __restrict__ a_dst, float* __restrict__ s,
              float* __restrict__ t)
{
  int wave = threadIdx.x >> 5, lane = threadIdx.x & 31;
  int row = blockIdx.x * 8 + wave;
  const float* hr = H + (size_t)row * Dc;
  float ds_ = 0.f, dt_ = 0.f;
  for (int e = lane; e < Dc; e += 32) { float hv = hr[e]; ds_ += hv * a_src[e]; dt_ += hv * a_dst[e]; }
  ds_ = wave_sum(ds_); dt_ = wave_sum(dt_);
  if (lane == 0) { s[row] = ds_; t[row] = dt_; }
}

// ---------------- GAT edge-softmax (3 passes) ----------------
DEV_INLINE void edge_endpoints(const int* __restrict__ ei, int i, int& srcI, int& tgtI) {
  if (i < Ec) { srcI = ei[i]; tgtI = ei[Ec + i]; }
  else        { srcI = tgtI = i - Ec; }           // self-loops appended
}
DEV_INLINE float edge_logit(const float* s, const float* t, int srcI, int tgtI) {
  float lg = s[srcI] + t[tgtI];
  return lg > 0.f ? lg : 0.2f * lg;               // leaky_relu(0.2)
}

__global__ void k_edge_max(const int* __restrict__ ei, const float* __restrict__ s,
                           const float* __restrict__ t, unsigned* __restrict__ mx)
{
  int i = blockIdx.x * 256 + threadIdx.x;
  if (i >= ENc) return;
  int srcI, tgtI; edge_endpoints(ei, i, srcI, tgtI);
  atomicMax(&mx[tgtI], fenc(edge_logit(s, t, srcI, tgtI)));
}

__global__ void k_edge_exp(const int* __restrict__ ei, const float* __restrict__ s,
                           const float* __restrict__ t, const unsigned* __restrict__ mx,
                           float* __restrict__ ex, float* __restrict__ den)
{
  int i = blockIdx.x * 256 + threadIdx.x;
  if (i >= ENc) return;
  int srcI, tgtI; edge_endpoints(ei, i, srcI, tgtI);
  float e_ = __expf(edge_logit(s, t, srcI, tgtI) - fdec(mx[tgtI]));
  ex[i] = e_;
  atomicAdd(&den[tgtI], e_);
}

// out[tgt,:] += (ex/den[tgt]) * H[src,:]  -- one thread per (edge, 4 dims)
__global__ void k_edge_agg(const int* __restrict__ ei, const float* __restrict__ ex,
                           const float* __restrict__ den, const float* __restrict__ H,
                           float* __restrict__ out)
{
  long long gid = (long long)blockIdx.x * 256 + threadIdx.x;
  if (gid >= (long long)ENc * 64) return;
  int i  = (int)(gid >> 6);
  int d4 = (int)(gid & 63) * 4;
  int srcI, tgtI; edge_endpoints(ei, i, srcI, tgtI);
  float alpha = ex[i] / den[tgtI];
  const float4 hv = *reinterpret_cast<const float4*>(H + (size_t)srcI * Dc + d4);
  float* o = out + (size_t)tgtI * Dc + d4;
  atomicAdd(o + 0, alpha * hv.x);
  atomicAdd(o + 1, alpha * hv.y);
  atomicAdd(o + 2, alpha * hv.z);
  atomicAdd(o + 3, alpha * hv.w);
}

// x = elu(agg + b); also f16 copy for the next WMMA GEMM
__global__ void k_bias_elu(float* __restrict__ X, h16* __restrict__ Xh,
                           const float* __restrict__ bias)
{
  int i = blockIdx.x * 256 + threadIdx.x;     // grid sized exactly Nn*Dc/256
  float v = X[i] + bias[i & (Dc - 1)];
  v = v > 0.f ? v : (__expf(v) - 1.f);
  X[i] = v; Xh[i] = (h16)v;
}

// ---------------- layernorm over rows of 256 ----------------
__global__ __launch_bounds__(256)
void k_ln(const float* __restrict__ R, const float* __restrict__ g,
          const float* __restrict__ b, float* __restrict__ X, h16* __restrict__ Xh)
{
  __shared__ float sm[8];
  int row = blockIdx.x, tdx = threadIdx.x;
  float v = R[(size_t)row * Dc + tdx];
  float ws = wave_sum(v);
  if ((tdx & 31) == 0) sm[tdx >> 5] = ws;
  __syncthreads();
  float tot = 0.f;
#pragma unroll
  for (int w = 0; w < 8; ++w) tot += sm[w];
  float mu = tot * (1.f / Dc);
  float d = v - mu;
  __syncthreads();
  float ws2 = wave_sum(d * d);
  if ((tdx & 31) == 0) sm[tdx >> 5] = ws2;
  __syncthreads();
  float tot2 = 0.f;
#pragma unroll
  for (int w = 0; w < 8; ++w) tot2 += sm[w];
  float y = d * rsqrtf(tot2 * (1.f / Dc) + 1e-5f) * g[tdx] + b[tdx];
  X[(size_t)row * Dc + tdx] = y;
  Xh[(size_t)row * Dc + tdx] = (h16)y;
}

// ---------------- fused attention: one wave per (batch, head, 16-row tile) ----------------
// V comes in TRANSPOSED ([D x Nn] f16) so P.V B-fragments are contiguous b128 loads.
__global__ __launch_bounds__(32)
void k_attn(const h16* __restrict__ Qh, const h16* __restrict__ Kh,
            const h16* __restrict__ Vt, const unsigned char* __restrict__ mask,
            h16* __restrict__ Oh)
{
  __shared__ __align__(16) float sS[16 * 256];   // fp32 scores / exp
  __shared__ __align__(16) h16   sP[16 * 256];   // f16 softmax probs
  int lane = threadIdx.x;
  int lt = blockIdx.x, head = blockIdx.y, bb = blockIdx.z;
  int rowbase = bb * Lc + lt * 16;
  int koff = head * HDc;
  int n = lane & 15, hh = lane >> 4;

  // S = Q K^T / sqrt(HD)  (K^T fragment = contiguous loads from row-major K)
  v16h aq[4];
#pragma unroll
  for (int kt = 0; kt < 4; ++kt) aq[kt] = load_a_frag(Qh, Dc, rowbase, koff + kt * 32);
  const float scale = 0.088388347648318447f;     // 1/sqrt(128)
  for (int nt = 0; nt < 16; ++nt) {
    v16h bk0 = load_b_frag_t(Kh, Dc, koff,      bb * Lc + nt * 16);
    v16h bk1 = load_b_frag_t(Kh, Dc, koff + 32, bb * Lc + nt * 16);
    v16h bk2 = load_b_frag_t(Kh, Dc, koff + 64, bb * Lc + nt * 16);
    v16h bk3 = load_b_frag_t(Kh, Dc, koff + 96, bb * Lc + nt * 16);
    v8f acc = {};
    acc = wmma_f16(aq[0], bk0, acc);
    acc = wmma_f16(aq[1], bk1, acc);
    acc = wmma_f16(aq[2], bk2, acc);
    acc = wmma_f16(aq[3], bk3, acc);
    int col = nt * 16 + n;
    float msk = mask[bb * Lc + col] ? 0.f : -1e9f;   // pad mask (all-valid here)
#pragma unroll
    for (int r = 0; r < 8; ++r) sS[(r + 8 * hh) * 256 + col] = acc[r] * scale + msk;
  }
  __syncthreads();

  // row-wise softmax, quantize to f16 probs
  for (int r = 0; r < 16; ++r) {
    float mx = -1e30f;
    for (int c = lane; c < 256; c += 32) mx = fmaxf(mx, sS[r * 256 + c]);
    mx = wave_max(mx);
    float sum = 0.f;
    for (int c = lane; c < 256; c += 32) {
      float e_ = __expf(sS[r * 256 + c] - mx);
      sS[r * 256 + c] = e_;
      sum += e_;
    }
    sum = wave_sum(sum);
    float inv = 1.f / sum;
    for (int c = lane; c < 256; c += 32) sP[r * 256 + c] = (h16)(sS[r * 256 + c] * inv);
  }
  __syncthreads();

  // O = P V   (P from LDS as A-fragments, V fragments contiguous from Vt)
  v16h ap[8];
#pragma unroll
  for (int kt = 0; kt < 8; ++kt) ap[kt] = load_a_frag(sP, 256, 0, kt * 32);
#pragma unroll
  for (int nt2 = 0; nt2 < 8; ++nt2) {
    v8f acc = {};
#pragma unroll
    for (int kt = 0; kt < 8; ++kt) {
      v16h bv = load_b_frag_t(Vt, Nn, bb * Lc + kt * 32, koff + nt2 * 16);
      acc = wmma_f16(ap[kt], bv, acc);
    }
#pragma unroll
    for (int r = 0; r < 8; ++r)
      Oh[(size_t)(rowbase + r + 8 * hh) * Dc + koff + nt2 * 16 + n] = (h16)acc[r];
  }
}

// ---------------- masked mean-pool + MLP head (tiny, VALU) ----------------
__global__ __launch_bounds__(256)
void k_head(const float* __restrict__ X, const unsigned char* __restrict__ mask,
            const float* __restrict__ W1, const float* __restrict__ b1,
            const float* __restrict__ W2, const float* __restrict__ b2,
            float* __restrict__ out)
{
  __shared__ float avg[256];
  __shared__ float sm[8];
  int bb = blockIdx.x, tdx = threadIdx.x;
  float s = 0.f, cnt = 0.f;
  for (int l = 0; l < Lc; ++l) {
    float mf = mask[bb * Lc + l] ? 1.f : 0.f;
    s += mf * X[(size_t)(bb * Lc + l) * Dc + tdx];
    cnt += mf;
  }
  avg[tdx] = s / cnt;
  __syncthreads();
  float hid = b1[tdx];
  for (int k = 0; k < Dc; ++k) hid += avg[k] * W1[k * Dc + tdx];
  hid = hid > 0.f ? hid : 0.f;
  float contrib = hid * W2[tdx];                 // W2 is [256,1]
  float ws = wave_sum(contrib);
  if ((tdx & 31) == 0) sm[tdx >> 5] = ws;
  __syncthreads();
  if (tdx == 0) {
    float tot = 0.f;
#pragma unroll
    for (int w = 0; w < 8; ++w) tot += sm[w];
    out[bb] = tot + b2[0];
  }
}

// ---------------- host orchestration ----------------
extern "C" void kernel_launch(void* const* d_in, const int* in_sizes, int n_in,
                              void* d_out, int out_size, void* d_ws, size_t ws_size,
                              hipStream_t stream)
{
  (void)in_sizes; (void)n_in; (void)out_size; (void)ws_size;
  const float* state_wf = (const float*)d_in[0];
  const int*   edge_index = (const int*)d_in[1];
  const unsigned char* mask_wf = (const unsigned char*)d_in[2];
  // d_in[3]=batch_wf (== i//L, dense layout assumed), d_in[4]=candidate_task_index (unused)

  // params: jax pytree leaves, dict keys sorted alphabetically
  struct AttnP { const float *W1,*W2,*Wk,*Wo,*Wq,*Wv,*b1,*b2,*bk,*bo,*bq,*bv,*ln1_b,*ln1_g,*ln2_b,*ln2_g; };
  struct GatP  { const float *W,*a_dst,*a_src,*b; };
  AttnP at[2]; GatP gp[2];
  int p = 5;
  for (int l = 0; l < 2; ++l) { const float** f = (const float**)&at[l]; for (int j = 0; j < 16; ++j) f[j] = (const float*)d_in[p++]; }
  for (int l = 0; l < 2; ++l) { const float** f = (const float**)&gp[l]; for (int j = 0; j <  4; ++j) f[j] = (const float*)d_in[p++]; }
  const float* mW1 = (const float*)d_in[p++];
  const float* mW2 = (const float*)d_in[p++];
  const float* mb1 = (const float*)d_in[p++];
  const float* mb2 = (const float*)d_in[p++];

  // workspace carve (~126 MB)
  char* wptr = (char*)d_ws;
  auto take = [&](size_t bytes) -> void* {
    void* r = wptr; wptr += (bytes + 255) & ~(size_t)255; return r;
  };
  h16*   sh   = (h16*)  take((size_t)Nn * INc * 2);   // state f16
  float* Hb   = (float*)take((size_t)Nn * Dc * 4);    // GAT h (pre-agg)
  float* Xb   = (float*)take((size_t)Nn * Dc * 4);    // activation fp32 / agg target
  float* Rb   = (float*)take((size_t)Nn * Dc * 4);    // pre-LN sum
  h16*   Xh   = (h16*)  take((size_t)Nn * Dc * 2);    // activation f16
  h16*   Fh   = (h16*)  take((size_t)Nn * FFc * 2);   // FFN intermediate f16
  h16*   Qh   = (h16*)  take((size_t)Nn * Dc * 2);
  h16*   Kh2  = (h16*)  take((size_t)Nn * Dc * 2);
  h16*   Vt   = (h16*)  take((size_t)Nn * Dc * 2);    // V stored transposed [D x Nn]
  h16*   Oh   = (h16*)  take((size_t)Nn * Dc * 2);
  float* sv   = (float*)take((size_t)Nn * 4);
  float* tv   = (float*)take((size_t)Nn * 4);
  unsigned* mxv = (unsigned*)take((size_t)Nn * 4);
  float* denv = (float*)take((size_t)Nn * 4);
  float* exv  = (float*)take((size_t)ENc * 4);
  h16* wg0t = (h16*)take((size_t)INc * Dc * 2);
  h16* wg1t = (h16*)take((size_t)Dc * Dc * 2);
  h16 *wqT[2], *wkT[2], *wvT[2], *woT[2], *w1T[2], *w2T[2];
  for (int l = 0; l < 2; ++l) {
    wqT[l] = (h16*)take((size_t)Dc * Dc * 2);
    wkT[l] = (h16*)take((size_t)Dc * Dc * 2);
    wvT[l] = (h16*)take((size_t)Dc * Dc * 2);
    woT[l] = (h16*)take((size_t)Dc * Dc * 2);
    w1T[l] = (h16*)take((size_t)Dc * FFc * 2);
    w2T[l] = (h16*)take((size_t)FFc * Dc * 2);
  }

  auto cvt  = [&](const float* s_, h16* d_, int nelem) {
    k_cvt<<<(nelem + 255) / 256, 256, 0, stream>>>(s_, d_, nelem);
  };
  auto cvtT = [&](const float* W_, h16* WT_, int K_, int N_) {
    k_cvt_t<<<(K_ * N_ + 255) / 256, 256, 0, stream>>>(W_, WT_, K_, N_);
  };

  // f16 conversions (weights transposed for contiguous B-fragments)
  cvt(state_wf, sh, Nn * INc);
  cvtT(gp[0].W, wg0t, INc, Dc);
  cvtT(gp[1].W, wg1t, Dc, Dc);
  for (int l = 0; l < 2; ++l) {
    cvtT(at[l].Wq, wqT[l], Dc, Dc);
    cvtT(at[l].Wk, wkT[l], Dc, Dc);
    cvtT(at[l].Wv, wvT[l], Dc, Dc);
    cvtT(at[l].Wo, woT[l], Dc, Dc);
    cvtT(at[l].W1, w1T[l], Dc, FFc);
    cvtT(at[l].W2, w2T[l], FFc, Dc);
  }

  dim3 gD(Nn / 128, Dc / 64);    // 16x64 strip per wave
  dim3 gF(Nn / 128, FFc / 64);

  // ---- GAT layers ----
  auto gat_layer = [&](const h16* Ain, int K_, const h16* WT_, const GatP& g) {
    k_gemm<<<gD, 256, 0, stream>>>(Ain, WT_, Nn, Dc, K_, nullptr, nullptr, Hb, nullptr, 0);
    k_rowdot<<<Nn / 8, 256, 0, stream>>>(Hb, g.a_src, g.a_dst, sv, tv);
    k_fill_u32<<<(Nn + 255) / 256, 256, 0, stream>>>(mxv, 0x007FFFFFu, Nn); // enc(-inf)
    hipMemsetAsync(denv, 0, (size_t)Nn * 4, stream);
    hipMemsetAsync(Xb, 0, (size_t)Nn * Dc * 4, stream);
    k_edge_max<<<(ENc + 255) / 256, 256, 0, stream>>>(edge_index, sv, tv, mxv);
    k_edge_exp<<<(ENc + 255) / 256, 256, 0, stream>>>(edge_index, sv, tv, mxv, exv, denv);
    long long tot = (long long)ENc * 64;
    k_edge_agg<<<(unsigned)((tot + 255) / 256), 256, 0, stream>>>(edge_index, exv, denv, Hb, Xb);
    k_bias_elu<<<(Nn * Dc) / 256, 256, 0, stream>>>(Xb, Xh, g.b);
  };
  gat_layer(sh, INc, wg0t, gp[0]);
  gat_layer(Xh, Dc, wg1t, gp[1]);

  // ---- transformer encoder layers (dense batch == reshape) ----
  for (int l = 0; l < 2; ++l) {
    k_gemm<<<gD, 256, 0, stream>>>(Xh, wqT[l], Nn, Dc, Dc, at[l].bq, nullptr, nullptr, Qh, 1);
    k_gemm<<<gD, 256, 0, stream>>>(Xh, wkT[l], Nn, Dc, Dc, at[l].bk, nullptr, nullptr, Kh2, 1);
    k_gemm<<<gD, 256, 0, stream>>>(Xh, wvT[l], Nn, Dc, Dc, at[l].bv, nullptr, nullptr, Vt, 4);
    dim3 ga(Lc / 16, NHc, Bc);
    k_attn<<<ga, 32, 0, stream>>>(Qh, Kh2, Vt, mask_wf, Oh);
    k_gemm<<<gD, 256, 0, stream>>>(Oh, woT[l], Nn, Dc, Dc, at[l].bo, Xb, Rb, nullptr, 3);
    k_ln<<<Nn, 256, 0, stream>>>(Rb, at[l].ln1_g, at[l].ln1_b, Xb, Xh);
    k_gemm<<<gF, 256, 0, stream>>>(Xh, w1T[l], Nn, FFc, Dc, at[l].b1, nullptr, nullptr, Fh, 2);
    k_gemm<<<gD, 256, 0, stream>>>(Fh, w2T[l], Nn, Dc, FFc, at[l].b2, Xb, Rb, nullptr, 3);
    k_ln<<<Nn, 256, 0, stream>>>(Rb, at[l].ln2_g, at[l].ln2_b, Xb, Xh);
  }

  // ---- pool + MLP head ----
  k_head<<<Bc, 256, 0, stream>>>(Xb, mask_wf, mW1, mb1, mW2, mb2, (float*)d_out);
}
